// NBodyGNN_58050777972751
// MI455X (gfx1250) — matrix operand order repaired and checked by reading, not compiled, not run on hardware
//
#include <hip/hip_runtime.h>
#include <hip/hip_bf16.h>

typedef long long i64;
typedef _Float16 f16;
typedef __attribute__((ext_vector_type(16))) _Float16 v16h;
typedef __attribute__((ext_vector_type(8)))  _Float16 v8h;
typedef __attribute__((ext_vector_type(8)))  float    v8f;

// ---------------------------------------------------------------------------
// Wave-level LDS fence: order cross-lane LDS RAW within a wave.
// ---------------------------------------------------------------------------
__device__ __forceinline__ void lds_fence() {
    __builtin_amdgcn_wave_barrier();
    asm volatile("s_wait_dscnt 0" ::: "memory");
    __builtin_amdgcn_wave_barrier();
}

// ---------------------------------------------------------------------------
// CDNA5 async global->LDS copy (ISA §15.18.3 op 98, ASYNCcnt-tracked).
// lds_off is the workgroup-relative LDS byte address (= low 32 bits of the
// generic pointer per the aperture rule, ISA §10.2).
// ---------------------------------------------------------------------------
__device__ __forceinline__ void async_gather_b128(unsigned lds_off, const void* gsrc) {
    asm volatile("global_load_async_to_lds_b128 %0, %1, off"
                 :: "v"(lds_off), "v"((unsigned long long)(uintptr_t)gsrc)
                 : "memory");
}
__device__ __forceinline__ void wait_async() {
    asm volatile("s_wait_asynccnt 0" ::: "memory");
}

// ---------------------------------------------------------------------------
// WMMA fragment loaders (CDNA5 16x16x32 f16, wave32 layouts per ISA 7.12.2)
// ---------------------------------------------------------------------------
__device__ __forceinline__ v16h a_frag(const f16* tile, int ldk, int kc, int lane) {
    int m  = lane & 15;
    int kb = (lane >> 4) * 8;
    const f16* p = tile + (size_t)m * ldk + kc * 32 + kb;
    v8h lo = *(const v8h*)(p);
    v8h hi = *(const v8h*)(p + 16);
    return __builtin_shufflevector(lo, hi, 0,1,2,3,4,5,6,7,8,9,10,11,12,13,14,15);
}

__device__ __forceinline__ v16h b_frag(const f16* Wt, int kpad, int ct, int kc, int lane) {
    int n  = lane & 15;
    int kh = lane >> 4;
    const f16* p = Wt + (size_t)(ct * 16 + n) * kpad + kc * 32 + kh * 8;
    v8h lo = *(const v8h*)(p);
    v8h hi = *(const v8h*)(p + 16);
    return __builtin_shufflevector(lo, hi, 0,1,2,3,4,5,6,7,8,9,10,11,12,13,14,15);
}

template<int NCT>
__device__ __forceinline__ void zero_acc(v8f (&acc)[NCT]) {
#pragma unroll
    for (int ct = 0; ct < NCT; ++ct)
#pragma unroll
        for (int r = 0; r < 8; ++r) acc[ct][r] = 0.f;
}

template<int MT, int NCT>
__device__ __forceinline__ void zero_acc_mt(v8f (&acc)[MT][NCT]) {
#pragma unroll
    for (int mt = 0; mt < MT; ++mt) zero_acc(acc[mt]);
}

// Single-M-tile GEMM (enc/dec)
template<int NCT>
__device__ __forceinline__ void gemm_acc(v8f (&acc)[NCT], const f16* A, int ldk, int nkc,
                                         const f16* Wt, int kpad, int lane) {
    for (int kc = 0; kc < nkc; ++kc) {
        v16h a = a_frag(A, ldk, kc, lane);
#pragma unroll
        for (int ct = 0; ct < NCT; ++ct) {
            v16h b = b_frag(Wt, kpad, ct, kc, lane);
            acc[ct] = __builtin_amdgcn_wmma_f32_16x16x32_f16(
                false, a, false, b, (short)0, acc[ct], false, false);
        }
    }
}

// M-blocked GEMM: one B-fragment load feeds MT WMMAs (halves weight traffic).
template<int MT, int NCT>
__device__ __forceinline__ void gemm_mt(v8f (&acc)[MT][NCT], const f16* const* A, int ldk,
                                        int nkc, const f16* Wt, int kpad, int lane) {
    for (int kc = 0; kc < nkc; ++kc) {
        v16h a[MT];
#pragma unroll
        for (int mt = 0; mt < MT; ++mt) a[mt] = a_frag(A[mt], ldk, kc, lane);
#pragma unroll
        for (int ct = 0; ct < NCT; ++ct) {
            v16h b = b_frag(Wt, kpad, ct, kc, lane);
#pragma unroll
            for (int mt = 0; mt < MT; ++mt)
                acc[mt][ct] = __builtin_amdgcn_wmma_f32_16x16x32_f16(
                    false, a[mt], false, b, (short)0, acc[mt][ct], false, false);
        }
    }
}

// C/D layout: element r -> row m = 8*(lane>>4) + r, col n = ct*16 + (lane&15)
template<int NCT>
__device__ __forceinline__ void dump_acc(const v8f (&acc)[NCT], const float* bias,
                                         float* sT, int lane) {
    int n  = lane & 15;
    int kh = lane >> 4;
#pragma unroll
    for (int ct = 0; ct < NCT; ++ct) {
        float bv = bias[ct * 16 + n];
#pragma unroll
        for (int r = 0; r < 8; ++r)
            sT[(8 * kh + r) * 128 + ct * 16 + n] = acc[ct][r] + bv;
    }
}

__device__ __forceinline__ float silu(float v) { return v * (1.f / (1.f + __expf(-v))); }

// LayerNorm(+SiLU) over a 16x128 f32 LDS tile -> 16x128 f16 LDS tile.
__device__ __forceinline__ void ln_silu_to_f16(const float* sT, const float* g, const float* bb,
                                               f16* out, int lane, bool act) {
    int m  = lane & 15;
    int kh = lane >> 4;
    const float* row = sT + m * 128;
    float s = 0.f, s2 = 0.f;
    for (int k = kh * 64; k < kh * 64 + 64; ++k) {
        float v = row[k];
        s += v; s2 += v * v;
    }
    s  += __shfl_xor(s, 16);
    s2 += __shfl_xor(s2, 16);
    float mean = s * (1.f / 128.f);
    float var  = s2 * (1.f / 128.f) - mean * mean;
    float rr   = rsqrtf(var + 1e-5f);
    for (int k = kh * 64; k < kh * 64 + 64; ++k) {
        float v = (row[k] - mean) * rr * g[k] + bb[k];
        if (act) v = silu(v);
        out[m * 128 + k] = (f16)v;
    }
}

// ---------------------------------------------------------------------------
// Prep kernels
// ---------------------------------------------------------------------------
__global__ void prep_w_kernel(const float* __restrict__ W, int K, int Nc, int kpad,
                              f16* __restrict__ Wt) {
    int n = blockIdx.x;
    for (int k = threadIdx.x; k < kpad; k += blockDim.x)
        Wt[(size_t)n * kpad + k] = (k < K) ? (f16)W[(size_t)k * Nc + n] : (f16)0.f;
}

__global__ void prep_edge_kernel(const i64* __restrict__ ei, const float* __restrict__ pos,
                                 f16* __restrict__ ea16, i64 E) {
    i64 e = (i64)blockIdx.x * blockDim.x + threadIdx.x;
    if (e >= E) return;
    i64 r = ei[e], c = ei[E + e];
    float dx = pos[c * 3 + 0] - pos[r * 3 + 0];
    float dy = pos[c * 3 + 1] - pos[r * 3 + 1];
    float dz = pos[c * 3 + 2] - pos[r * 3 + 2];
    float d  = sqrtf(dx * dx + dy * dy + dz * dz) + 1e-8f;
    float inv = 1.f / d;
    ea16[e * 4 + 0] = (f16)d;
    ea16[e * 4 + 1] = (f16)(dx * inv);
    ea16[e * 4 + 2] = (f16)(dy * inv);
    ea16[e * 4 + 3] = (f16)(dz * inv);
}

__global__ void zero_f32_kernel(float* __restrict__ p, i64 n) {
    i64 i = (i64)blockIdx.x * blockDim.x + threadIdx.x;
    if (i < n) p[i] = 0.f;
}

// ---------------------------------------------------------------------------
// Encoder: h = l2( silu( LN( l1(x) ) ) )
// ---------------------------------------------------------------------------
__global__ void __launch_bounds__(128)
enc_kernel(const float* __restrict__ x, const float* __restrict__ W1,
           const float* __restrict__ b1, const float* __restrict__ lng,
           const float* __restrict__ lnb, const f16* __restrict__ W2t,
           const float* __restrict__ b2, float* __restrict__ h,
           f16* __restrict__ h16, i64 N, i64 nTiles) {
    __shared__ __align__(16) float sT[4][16 * 128];
    __shared__ __align__(16) f16   A16[4][16 * 128];
    int wid = threadIdx.x >> 5, lane = threadIdx.x & 31;
    int m = lane & 15, kh = lane >> 4, n = lane & 15;
    float* st = sT[wid];
    f16*   a16 = A16[wid];
    for (i64 tile = (i64)blockIdx.x * 4 + wid; tile < nTiles; tile += (i64)gridDim.x * 4) {
        i64 n0 = tile * 16;
        i64 nr = n0 + m; if (nr >= N) nr = N - 1;
        float xr[9];
#pragma unroll
        for (int k = 0; k < 9; ++k) xr[k] = x[nr * 9 + k];
        for (int c = kh * 64; c < kh * 64 + 64; ++c) {
            float s = b1[c];
#pragma unroll
            for (int k = 0; k < 9; ++k) s += xr[k] * W1[k * 128 + c];
            st[m * 128 + c] = s;
        }
        lds_fence();
        ln_silu_to_f16(st, lng, lnb, a16, lane, true);
        lds_fence();
        v8f acc[8]; zero_acc(acc);
        gemm_acc(acc, a16, 128, 4, W2t, 128, lane);
#pragma unroll
        for (int ct = 0; ct < 8; ++ct) {
            float bv = b2[ct * 16 + n];
#pragma unroll
            for (int r = 0; r < 8; ++r) {
                i64 row = n0 + 8 * kh + r;
                if (row < N) {
                    float v = acc[ct][r] + bv;
                    h[row * 128 + ct * 16 + n]   = v;
                    h16[row * 128 + ct * 16 + n] = (f16)v;
                }
            }
        }
        lds_fence();
    }
}

// ---------------------------------------------------------------------------
// Edge kernel, MT=2: one wave owns 32 edges (two 16-row A-tiles).
// x_i/x_j gathers go straight to LDS via async b128 copies (ASYNCcnt).
// ---------------------------------------------------------------------------
__global__ void __launch_bounds__(128)
edge_kernel(const i64* __restrict__ ei, const f16* __restrict__ h16,
            const f16* __restrict__ ea16,
            const f16* __restrict__ W1i, const f16* __restrict__ W1j,
            const f16* __restrict__ W1e, const float* __restrict__ b1,
            const float* __restrict__ ln1g, const float* __restrict__ ln1b,
            const f16* __restrict__ W2t, const float* __restrict__ b2,
            const float* __restrict__ ln2g, const float* __restrict__ ln2b,
            const f16* __restrict__ W3t, const float* __restrict__ b3,
            float* __restrict__ aggr, i64 E, i64 nTiles) {
    __shared__ __align__(16) f16   Ati[4][2][16 * 128];
    __shared__ __align__(16) f16   Atj[4][2][16 * 128];
    __shared__ __align__(16) f16   Aet[4][2][16 * 32];
    __shared__ __align__(16) float sTt[4][2][16 * 128];
    int wid = threadIdx.x >> 5, lane = threadIdx.x & 31;
    int m = lane & 15, kh = lane >> 4;
    for (i64 tile = (i64)blockIdx.x * 4 + wid; tile < nTiles; tile += (i64)gridDim.x * 4) {
        i64 e0 = tile * 32;
        i64 dst[2];
        bool valid[2];
#pragma unroll
        for (int mt = 0; mt < 2; ++mt) {
            i64 e = e0 + mt * 16 + m;
            valid[mt] = (e < E);
            if (!valid[mt]) e = E - 1;
            i64 src = ei[e];
            dst[mt] = ei[E + e];
            // edge-attr tile: 16x32 zero-padded
            f16* ae = Aet[wid][mt];
            for (int t = 0; t < 16; ++t) ae[m * 32 + kh * 16 + t] = (f16)0.f;
            if (kh == 0) {
#pragma unroll
                for (int t = 0; t < 4; ++t) ae[m * 32 + t] = ea16[e * 4 + t];
            }
            // async gathers: this lane owns half of row m (128B) in each tile
            unsigned li = (unsigned)(uintptr_t)(&Ati[wid][mt][m * 128 + kh * 64]);
            unsigned lj = (unsigned)(uintptr_t)(&Atj[wid][mt][m * 128 + kh * 64]);
            const f16* gi = h16 + dst[mt] * 128 + kh * 64;
            const f16* gj = h16 + src * 128 + kh * 64;
#pragma unroll
            for (int t = 0; t < 8; ++t) {
                async_gather_b128(li + t * 16, gi + t * 8);
                async_gather_b128(lj + t * 16, gj + t * 8);
            }
        }
        wait_async();
        lds_fence();
        const f16* Ai[2] = {Ati[wid][0], Ati[wid][1]};
        const f16* Aj[2] = {Atj[wid][0], Atj[wid][1]};
        const f16* Ae[2] = {Aet[wid][0], Aet[wid][1]};
        v8f acc[2][8];
        zero_acc_mt(acc);
        gemm_mt(acc, Ai, 128, 4, W1i, 128, lane);   // x_i part
        gemm_mt(acc, Aj, 128, 4, W1j, 128, lane);   // x_j part
        gemm_mt(acc, Ae,  32, 1, W1e,  32, lane);   // edge-attr part (K padded to 32)
#pragma unroll
        for (int mt = 0; mt < 2; ++mt) dump_acc(acc[mt], b1, sTt[wid][mt], lane);
        lds_fence();
#pragma unroll
        for (int mt = 0; mt < 2; ++mt)
            ln_silu_to_f16(sTt[wid][mt], ln1g, ln1b, Ati[wid][mt], lane, true);
        lds_fence();
        zero_acc_mt(acc);
        gemm_mt(acc, Ai, 128, 4, W2t, 128, lane);
#pragma unroll
        for (int mt = 0; mt < 2; ++mt) dump_acc(acc[mt], b2, sTt[wid][mt], lane);
        lds_fence();
#pragma unroll
        for (int mt = 0; mt < 2; ++mt)
            ln_silu_to_f16(sTt[wid][mt], ln2g, ln2b, Ati[wid][mt], lane, true);
        lds_fence();
        zero_acc_mt(acc);
        gemm_mt(acc, Ai, 128, 4, W3t, 128, lane);
#pragma unroll
        for (int mt = 0; mt < 2; ++mt) dump_acc(acc[mt], b3, sTt[wid][mt], lane);
        lds_fence();
#pragma unroll
        for (int mt = 0; mt < 2; ++mt) {
            if (valid[mt]) {
                float* dp = aggr + dst[mt] * 128 + kh * 64;
                const float* sp = sTt[wid][mt] + m * 128 + kh * 64;
                for (int t = 0; t < 64; ++t)
                    __hip_atomic_fetch_add(dp + t, sp[t], __ATOMIC_RELAXED,
                                           __HIP_MEMORY_SCOPE_AGENT);
            }
        }
        lds_fence();
    }
}

// ---------------------------------------------------------------------------
// Node kernel, MT=2: h_new = l2(silu(LN(l1([h,aggr])))); h = LN_norm(h+h_new)
// ---------------------------------------------------------------------------
__global__ void __launch_bounds__(128)
node_kernel(float* __restrict__ h, f16* __restrict__ h16, const float* __restrict__ aggr,
            const f16* __restrict__ W1h, const f16* __restrict__ W1a,
            const float* __restrict__ b1, const float* __restrict__ lng,
            const float* __restrict__ lnb, const f16* __restrict__ W2t,
            const float* __restrict__ b2, const float* __restrict__ nmg,
            const float* __restrict__ nmb, i64 N, i64 nTiles) {
    __shared__ __align__(16) f16   Aat[4][2][16 * 128];
    __shared__ __align__(16) f16   A16t[4][2][16 * 128];
    __shared__ __align__(16) float sTt[4][2][16 * 128];
    int wid = threadIdx.x >> 5, lane = threadIdx.x & 31;
    int m = lane & 15, kh = lane >> 4;
    for (i64 tile = (i64)blockIdx.x * 4 + wid; tile < nTiles; tile += (i64)gridDim.x * 4) {
        i64 n0 = tile * 32;
#pragma unroll
        for (int mt = 0; mt < 2; ++mt) {
            f16* aa = Aat[wid][mt];
            i64 row = n0 + mt * 16 + m;
            for (int k = kh * 64; k < kh * 64 + 64; ++k)
                aa[m * 128 + k] = (f16)aggr[row * 128 + k];
        }
        lds_fence();
        const f16* Ah[2]  = {h16 + n0 * 128, h16 + (n0 + 16) * 128};
        const f16* Aa[2]  = {Aat[wid][0], Aat[wid][1]};
        const f16* Ax[2]  = {A16t[wid][0], A16t[wid][1]};
        v8f acc[2][8];
        zero_acc_mt(acc);
        gemm_mt(acc, Ah, 128, 4, W1h, 128, lane);   // h rows straight from global
        gemm_mt(acc, Aa, 128, 4, W1a, 128, lane);
#pragma unroll
        for (int mt = 0; mt < 2; ++mt) dump_acc(acc[mt], b1, sTt[wid][mt], lane);
        lds_fence();
#pragma unroll
        for (int mt = 0; mt < 2; ++mt)
            ln_silu_to_f16(sTt[wid][mt], lng, lnb, A16t[wid][mt], lane, true);
        lds_fence();
        zero_acc_mt(acc);
        gemm_mt(acc, Ax, 128, 4, W2t, 128, lane);
#pragma unroll
        for (int mt = 0; mt < 2; ++mt) dump_acc(acc[mt], b2, sTt[wid][mt], lane);
        lds_fence();
#pragma unroll
        for (int mt = 0; mt < 2; ++mt) {
            float* st = sTt[wid][mt];
            i64 row = n0 + mt * 16 + m;
            float s = 0.f, s2 = 0.f;
            for (int k = kh * 64; k < kh * 64 + 64; ++k) {
                float v = h[row * 128 + k] + st[m * 128 + k];
                st[m * 128 + k] = v;
                s += v; s2 += v * v;
            }
            s  += __shfl_xor(s, 16);
            s2 += __shfl_xor(s2, 16);
            float mean = s * (1.f / 128.f);
            float var  = s2 * (1.f / 128.f) - mean * mean;
            float rr   = rsqrtf(var + 1e-5f);
            if (row < N) {
                for (int k = kh * 64; k < kh * 64 + 64; ++k) {
                    float v = (st[m * 128 + k] - mean) * rr * nmg[k] + nmb[k];
                    h[row * 128 + k]   = v;
                    h16[row * 128 + k] = (f16)v;
                }
            }
        }
        lds_fence();
    }
}

// ---------------------------------------------------------------------------
// Decoder: out = l3( silu(l2( silu(l1(h)) )) )
// ---------------------------------------------------------------------------
__global__ void __launch_bounds__(128)
dec_kernel(const f16* __restrict__ h16, const f16* __restrict__ W1t,
           const float* __restrict__ b1, const f16* __restrict__ W2t,
           const float* __restrict__ b2, const float* __restrict__ W3,
           const float* __restrict__ b3, float* __restrict__ out,
           i64 N, i64 nTiles) {
    __shared__ __align__(16) f16 A16[4][16 * 128];
    __shared__ __align__(16) f16 A2[4][16 * 64];
    int wid = threadIdx.x >> 5, lane = threadIdx.x & 31;
    int m = lane & 15, kh = lane >> 4, n = lane & 15;
    f16* a16 = A16[wid];
    f16* a2  = A2[wid];
    for (i64 tile = (i64)blockIdx.x * 4 + wid; tile < nTiles; tile += (i64)gridDim.x * 4) {
        i64 n0 = tile * 16;
        v8f acc[8]; zero_acc(acc);
        gemm_acc(acc, h16 + n0 * 128, 128, 4, W1t, 128, lane);
#pragma unroll
        for (int ct = 0; ct < 8; ++ct) {
            float bv = b1[ct * 16 + n];
#pragma unroll
            for (int r = 0; r < 8; ++r)
                a16[(8 * kh + r) * 128 + ct * 16 + n] = (f16)silu(acc[ct][r] + bv);
        }
        lds_fence();
        v8f acc2[4]; zero_acc(acc2);
        gemm_acc(acc2, a16, 128, 4, W2t, 128, lane);
#pragma unroll
        for (int ct = 0; ct < 4; ++ct) {
            float bv = b2[ct * 16 + n];
#pragma unroll
            for (int r = 0; r < 8; ++r)
                a2[(8 * kh + r) * 64 + ct * 16 + n] = (f16)silu(acc2[ct][r] + bv);
        }
        lds_fence();
        if (kh == 0) {
            i64 row = n0 + m;
            if (row < N) {
#pragma unroll
                for (int o = 0; o < 6; ++o) {
                    float s = b3[o];
                    for (int k = 0; k < 64; ++k)
                        s += (float)a2[m * 64 + k] * W3[k * 6 + o];
                    out[row * 6 + o] = s;
                }
            }
        }
        lds_fence();
    }
}

// ---------------------------------------------------------------------------
// Host orchestration
// ---------------------------------------------------------------------------
enum {
    DEC_L1_B = 0, DEC_L1_W, DEC_L2_B, DEC_L2_W, DEC_L3_B, DEC_L3_W,
    ENC_L1_B, ENC_L1_W, ENC_L2_B, ENC_L2_W, ENC_LN_B, ENC_LN_G,
    LAYER0 = 12
};
enum {
    E_L1_B = 0, E_L1_W, E_L2_B, E_L2_W, E_L3_B, E_L3_W,
    E_LN1_B, E_LN1_G, E_LN2_B, E_LN2_G,
    N_L1_B, N_L1_W, N_L2_B, N_L2_W, N_LN_B, N_LN_G,
    NORM_B, NORM_G, LAYER_STRIDE = 18
};

extern "C" void kernel_launch(void* const* d_in, const int* in_sizes, int n_in,
                              void* d_out, int out_size, void* d_ws, size_t ws_size,
                              hipStream_t stream) {
    (void)out_size; (void)ws_size;
    const float* x   = (const float*)d_in[0];
    const i64*   ei  = (const i64*)d_in[1];
    const float* pos = (const float*)d_in[2];
    i64 N = in_sizes[0] / 9;
    i64 E = in_sizes[1] / 2;

    const float* leaf[84];
    static const int head_sizes[12]  = {128, 16384, 64, 8192, 6, 384,
                                        128, 1152, 128, 16384, 128, 128};
    static const int layer_sizes[18] = {128, 33280, 128, 16384, 128, 16384,
                                        128, 128, 128, 128,
                                        128, 32768, 128, 16384, 128, 128,
                                        128, 128};
    if (n_in >= 3 + 84) {
        for (int i = 0; i < 84; ++i) leaf[i] = (const float*)d_in[3 + i];
    } else {
        const float* base = (const float*)d_in[3];
        long long off = 0;
        for (int i = 0; i < 12; ++i) { leaf[i] = base + off; off += head_sizes[i]; }
        for (int l = 0; l < 4; ++l)
            for (int i = 0; i < 18; ++i) {
                leaf[LAYER0 + LAYER_STRIDE * l + i] = base + off;
                off += layer_sizes[i];
            }
    }

    char* wsp = (char*)d_ws;
    size_t off = 0;
    auto alloc = [&](size_t bytes) -> void* {
        void* p = wsp + off;
        off = (off + bytes + 255) & ~(size_t)255;
        return p;
    };
    float* h      = (float*)alloc((size_t)N * 128 * 4);
    f16*   h16    = (f16*)  alloc((size_t)N * 128 * 2);
    float* aggr   = (float*)alloc((size_t)N * 128 * 4);
    f16*   ea16   = (f16*)  alloc((size_t)E * 4 * 2);
    f16*   encl2t = (f16*)  alloc(128 * 128 * 2);
    f16 *eW1i[4], *eW1j[4], *eW1e[4], *eW2[4], *eW3[4], *nW1h[4], *nW1a[4], *nW2[4];
    for (int l = 0; l < 4; ++l) {
        eW1i[l] = (f16*)alloc(128 * 128 * 2);
        eW1j[l] = (f16*)alloc(128 * 128 * 2);
        eW1e[l] = (f16*)alloc(128 * 32 * 2);
        eW2[l]  = (f16*)alloc(128 * 128 * 2);
        eW3[l]  = (f16*)alloc(128 * 128 * 2);
        nW1h[l] = (f16*)alloc(128 * 128 * 2);
        nW1a[l] = (f16*)alloc(128 * 128 * 2);
        nW2[l]  = (f16*)alloc(128 * 128 * 2);
    }
    f16* decl1t = (f16*)alloc(128 * 128 * 2);
    f16* decl2t = (f16*)alloc(64 * 128 * 2);

    auto prep = [&](const float* W, int K, int Nc, int kpad, f16* Wt) {
        prep_w_kernel<<<Nc, 128, 0, stream>>>(W, K, Nc, kpad, Wt);
    };
    prep(leaf[ENC_L2_W], 128, 128, 128, encl2t);
    for (int l = 0; l < 4; ++l) {
        const int b = LAYER0 + LAYER_STRIDE * l;
        prep(leaf[b + E_L1_W],             128, 128, 128, eW1i[l]);  // rows   0..127 (x_i)
        prep(leaf[b + E_L1_W] + 128 * 128, 128, 128, 128, eW1j[l]);  // rows 128..255 (x_j)
        prep(leaf[b + E_L1_W] + 256 * 128,   4, 128,  32, eW1e[l]);  // rows 256..259 (ea)
        prep(leaf[b + E_L2_W], 128, 128, 128, eW2[l]);
        prep(leaf[b + E_L3_W], 128, 128, 128, eW3[l]);
        prep(leaf[b + N_L1_W],             128, 128, 128, nW1h[l]);  // rows   0..127 (h)
        prep(leaf[b + N_L1_W] + 128 * 128, 128, 128, 128, nW1a[l]);  // rows 128..255 (aggr)
        prep(leaf[b + N_L2_W], 128, 128, 128, nW2[l]);
    }
    prep(leaf[DEC_L1_W], 128, 128, 128, decl1t);
    prep(leaf[DEC_L2_W], 128,  64, 128, decl2t);

    prep_edge_kernel<<<(unsigned)((E + 255) / 256), 256, 0, stream>>>(ei, pos, ea16, E);

    i64 nTiles16 = (N + 15) / 16;
    i64 nTilesN32 = (N + 31) / 32;
    unsigned nBlocks16 = (unsigned)((nTiles16 + 3) / 4);
    unsigned nBlocksN32 = (unsigned)((nTilesN32 + 3) / 4);
    enc_kernel<<<nBlocks16, 128, 0, stream>>>(
        x, leaf[ENC_L1_W], leaf[ENC_L1_B], leaf[ENC_LN_G], leaf[ENC_LN_B],
        encl2t, leaf[ENC_L2_B], h, h16, N, nTiles16);

    i64 nTilesE32 = (E + 31) / 32;
    unsigned nBlocksE = (unsigned)((nTilesE32 + 3) / 4);
    for (int l = 0; l < 4; ++l) {
        const int b = LAYER0 + LAYER_STRIDE * l;
        zero_f32_kernel<<<(unsigned)((N * 128 + 255) / 256), 256, 0, stream>>>(aggr, N * 128);
        edge_kernel<<<nBlocksE, 128, 0, stream>>>(
            ei, h16, ea16,
            eW1i[l], eW1j[l], eW1e[l], leaf[b + E_L1_B],
            leaf[b + E_LN1_G], leaf[b + E_LN1_B],
            eW2[l], leaf[b + E_L2_B], leaf[b + E_LN2_G], leaf[b + E_LN2_B],
            eW3[l], leaf[b + E_L3_B],
            aggr, E, nTilesE32);
        node_kernel<<<nBlocksN32, 128, 0, stream>>>(
            h, h16, aggr,
            nW1h[l], nW1a[l], leaf[b + N_L1_B],
            leaf[b + N_LN_G], leaf[b + N_LN_B],
            nW2[l], leaf[b + N_L2_B],
            leaf[b + NORM_G], leaf[b + NORM_B],
            N, nTilesN32);
    }

    dec_kernel<<<nBlocks16, 128, 0, stream>>>(
        h16, decl1t, leaf[DEC_L1_B], decl2t, leaf[DEC_L2_B],
        leaf[DEC_L3_W], leaf[DEC_L3_B], (float*)d_out, N, nTiles16);
}
